// SBF_nlfc_IBF_nlfc_34754875359798
// MI455X (gfx1250) — matrix operand order repaired and checked by reading, not compiled, not run on hardware
//
#include <hip/hip_runtime.h>
#include <hip/hip_bf16.h>

// ---------------------------------------------------------------------------
// Types
// ---------------------------------------------------------------------------
typedef __bf16 bf16_t;
typedef bf16_t v16bf __attribute__((ext_vector_type(16)));
typedef bf16_t bf16x8 __attribute__((ext_vector_type(8)));
typedef float  v8f   __attribute__((ext_vector_type(8)));
typedef float  f32x4 __attribute__((ext_vector_type(4)));

// ---------------------------------------------------------------------------
// Problem constants
// ---------------------------------------------------------------------------
#define CCH   1024      // channels
#define ICH   512       // inter channels
#define BB    64        // batch
#define NPOS  784       // T*H*W = 4*14*14
#define NP2   196       // pooled positions 4*7*7
#define NP2P  224       // padded to multiple of 32
#define BCHUNK 8        // batches per chunk

// GEMM tiling: 256 threads = 8 waves (wave32); wave grid 4x2; wave tile 32x64
#define BLK_M 128
#define BLK_N 128
#define BLK_K 32
#define LDA_S 48        // LDS stride (elems) for A tile rows: 96B (16B aligned)
#define LDB_SW 40       // LDS stride (elems) for BT rows (n-major): 80B (16B aligned)
#define NBUF  3         // async pipeline depth

// ---------------------------------------------------------------------------
// Async global->LDS 16B copies (CDNA5, ASYNCcnt). INST_OFFSET applies to both
// the LDS destination and the global source address.
// ---------------------------------------------------------------------------
__device__ __forceinline__ void async_copy16(unsigned lds_off, const void* gaddr) {
  asm volatile("global_load_async_to_lds_b128 %0, %1, off"
               :: "v"(lds_off), "v"(gaddr) : "memory");
}
__device__ __forceinline__ void async_copy16_o16(unsigned lds_off, const void* gaddr) {
  asm volatile("global_load_async_to_lds_b128 %0, %1, off offset:16"
               :: "v"(lds_off), "v"(gaddr) : "memory");
}
__device__ __forceinline__ void wait_async0() {
  asm volatile("s_wait_asynccnt 0x0" ::: "memory");
}
// Each stage issues exactly 4 async ops per lane; in-order retirement means
// "<= 4 outstanding" == "everything except the newest tile has landed".
__device__ __forceinline__ void wait_async_le4() {
  asm volatile("s_wait_asynccnt 0x4" ::: "memory");
}

// ---------------------------------------------------------------------------
// WMMA fragment loads (CDNA5 ISA, wave32)
// A 16x32 bf16: lanes 0-15 row M=lane, K in {0..7,16..23}; lanes 16-31 K+8
// ---------------------------------------------------------------------------
__device__ inline v16bf load_frag_a(const bf16_t* As, int mt, int lane) {
  const int lr = lane & 15, lh = lane >> 4;
  const bf16_t* p = As + (mt + lr) * LDA_S + lh * 8;
  v16bf a;
#pragma unroll
  for (int j = 0; j < 8; ++j) a[j] = p[j];          // K = lh*8 + j
#pragma unroll
  for (int j = 8; j < 16; ++j) a[j] = p[j + 8];     // K = 16 + lh*8 + (j-8)
  return a;
}

// B 32x16 bf16: lane holds column n=lane&15; lanes 0-15 K in {0..7,16..23}; lanes 16-31 K+8.
// LDS holds B transposed (n-major, k-contiguous) -> 2x ds_load_b128 per lane.
__device__ inline v16bf load_frag_b(const bf16_t* Bs, int nt, int lane) {
  const int lr = lane & 15, lh = lane >> 4;
  const bf16_t* p = Bs + (nt + lr) * LDB_SW + lh * 8;
  v16bf b;
#pragma unroll
  for (int j = 0; j < 16; ++j) b[j] = p[j + (j >> 3) * 8];
  return b;
}

// ---------------------------------------------------------------------------
// Triple-buffered async GEMM main loop: acc[2][4] += A(m0:,k) * BT(n0:,k)^T
// A : bf16 row-major (M x K); BT: bf16 row-major (N x K). K % 32 == 0,
// K >= 64, full K extent addressable (callers guarantee).
// Steady-state body is branch-free; last two iterations are peeled.
// ---------------------------------------------------------------------------
__device__ inline void gemm_mainloop(const bf16_t* __restrict__ A,
                                     const bf16_t* __restrict__ BT,
                                     int M, int Nn, int K, int lda, int ldb,
                                     int m0, int n0,
                                     bf16_t* As, bf16_t* Bs, v8f acc[2][4]) {
  const int tid  = threadIdx.x;
  const int lane = tid & 31;
  const int wave = tid >> 5;
  const int wm = wave >> 1, wn = wave & 1;

  // A staging: row = tid>>1 (0..127), 32B half = tid&1; branchless row clamp
  const int rowA = tid >> 1;
  const int gmA  = m0 + rowA;
  const int gmAc = (gmA < M) ? gmA : (M - 1);
  const unsigned offA = (unsigned)(rowA * (LDA_S * 2) + (tid & 1) * 32);
  // BT staging: row = tid>>1 (0..127), 32B half = tid&1
  const int gnB  = n0 + rowA;
  const int gnBc = (gnB < Nn) ? gnB : (Nn - 1);
  const unsigned offB = (unsigned)(rowA * (LDB_SW * 2) + (tid & 1) * 32);

  const unsigned baseA = (unsigned)(size_t)As;
  const unsigned baseB = (unsigned)(size_t)Bs;
  const unsigned bufAbytes = BLK_M * LDA_S * 2;
  const unsigned bufBbytes = BLK_N * LDB_SW * 2;

  auto stage = [&](int buf, int k0) {
    const void* ga = (const void*)(A + (size_t)gmAc * lda + (k0 + (tid & 1) * 16));
    const unsigned la = baseA + (unsigned)buf * bufAbytes + offA;
    async_copy16(la, ga);
    async_copy16_o16(la, ga);
    const void* gb = (const void*)(BT + (size_t)gnBc * ldb + (k0 + (tid & 1) * 16));
    const unsigned lb = baseB + (unsigned)buf * bufBbytes + offB;
    async_copy16(lb, gb);
    async_copy16_o16(lb, gb);
  };

  auto compute = [&](int cur) {
    const bf16_t* Ac = As + cur * (BLK_M * LDA_S);
    const bf16_t* Bc = Bs + cur * (BLK_N * LDB_SW);
    v16bf a0 = load_frag_a(Ac, wm * 32,      lane);
    v16bf a1 = load_frag_a(Ac, wm * 32 + 16, lane);
#pragma unroll
    for (int j = 0; j < 4; ++j) {
      v16bf bj = load_frag_b(Bc, wn * 64 + j * 16, lane);
      acc[0][j] = __builtin_amdgcn_wmma_f32_16x16x32_bf16(false, a0, false, bj, (short)0, acc[0][j], false, false);
      acc[1][j] = __builtin_amdgcn_wmma_f32_16x16x32_bf16(false, a1, false, bj, (short)0, acc[1][j], false, false);
    }
  };

  const int niter = K / BLK_K;   // >= 2 for all callers
  stage(0, 0);
  stage(1, BLK_K);
  wait_async_le4();              // tile 0 resident; tile 1 may be in flight
  __syncthreads();
  int bufsel = 0;
  for (int it = 0; it < niter - 2; ++it) {   // branch-free steady state
    int nb = bufsel + 2; if (nb >= NBUF) nb -= NBUF;
    stage(nb, (it + 2) * BLK_K);
    compute(bufsel);
    wait_async_le4();
    __syncthreads();
    bufsel = (bufsel + 1 == NBUF) ? 0 : bufsel + 1;
  }
  compute(bufsel);               // tile niter-2
  wait_async0();                 // drain: last tile resident
  __syncthreads();
  bufsel = (bufsel + 1 == NBUF) ? 0 : bufsel + 1;
  compute(bufsel);               // tile niter-1
}

// ---------------------------------------------------------------------------
// Generic WMMA GEMM kernel.
// OMODE: 0 = f32 store, 1 = bf16 store, 2 = bf16 transposed store D[n,m]
// BMODE: 0 = none, 1 = bias[m] (row/out-channel), 2 = bias[n] (col)
// ---------------------------------------------------------------------------
struct GemmP {
  const bf16_t* A; const bf16_t* BT; const float* bias; void* D;
  long sA, sB, sD;            // per-batch strides (elements)
  int lda, ldb, ldd, M, N, K;
};

template<int OMODE, int BMODE>
__global__ __launch_bounds__(256) void gemm_bf16_kernel(GemmP p) {
  __shared__ bf16_t As[NBUF * BLK_M * LDA_S] __attribute__((aligned(16)));
  __shared__ bf16_t Bs[NBUF * BLK_N * LDB_SW] __attribute__((aligned(16)));
  const int z  = blockIdx.z;
  const int m0 = blockIdx.y * BLK_M;
  const int n0 = blockIdx.x * BLK_N;
  const bf16_t* A  = p.A  + (size_t)z * p.sA;
  const bf16_t* BT = p.BT + (size_t)z * p.sB;
  v8f acc[2][4] = {};
  gemm_mainloop(A, BT, p.M, p.N, p.K, p.lda, p.ldb, m0, n0, As, Bs, acc);

  const int tid = threadIdx.x, lane = tid & 31, wave = tid >> 5;
  const int wm = wave >> 1, wn = wave & 1;
  const int lr = lane & 15, lh = lane >> 4;
#pragma unroll
  for (int i = 0; i < 2; ++i) {
#pragma unroll
    for (int j = 0; j < 4; ++j) {
      const int n = n0 + wn * 64 + j * 16 + lr;
      if (n >= p.N) continue;
      const int mb = m0 + wm * 32 + i * 16 + lh * 8;
      if (OMODE == 2 && mb + 8 <= p.M) {
        // transposed store: 8 consecutive m per lane -> one 16B store
        bf16x8 o;
#pragma unroll
        for (int r = 0; r < 8; ++r) {
          float v = acc[i][j][r];
          if (BMODE == 1) v += p.bias[mb + r];
          if (BMODE == 2) v += p.bias[n];
          o[r] = (bf16_t)v;
        }
        *(bf16x8*)((bf16_t*)p.D + (size_t)z * p.sD + (size_t)n * p.ldd + mb) = o;
        continue;
      }
#pragma unroll
      for (int r = 0; r < 8; ++r) {
        const int m = mb + r;
        if (m >= p.M) continue;
        float v = acc[i][j][r];
        if (BMODE == 1) v += p.bias[m];
        if (BMODE == 2) v += p.bias[n];
        if (OMODE == 0) {
          ((float*)p.D)[(size_t)z * p.sD + (size_t)m * p.ldd + n] = v;
        } else if (OMODE == 1) {
          ((bf16_t*)p.D)[(size_t)z * p.sD + (size_t)m * p.ldd + n] = (bf16_t)v;
        } else {
          ((bf16_t*)p.D)[(size_t)z * p.sD + (size_t)n * p.ldd + m] = (bf16_t)v;
        }
      }
    }
  }
}

// ---------------------------------------------------------------------------
// Output-projection GEMM with fused bias + BatchNorm + residual + mean-pool.
// Rows m = spatial positions (784), cols n = channels (1024).
// pool[b, c] += (1/784) * sum_m ((acc+w_b - bn_m)*inv + bn_b + x[b,c,m])
// ---------------------------------------------------------------------------
struct GemmZP {
  const bf16_t* A; const bf16_t* BT;   // BT = w_w bf16 (C x IC)
  const float* w_b; const float* bn_g; const float* bn_b;
  const float* bn_m; const float* bn_v;
  const float* x; float* pool;
  long sA; int lda, ldb, M, K, c0;
};

__global__ __launch_bounds__(256) void gemm_z_kernel(GemmZP p) {
  __shared__ bf16_t As[NBUF * BLK_M * LDA_S] __attribute__((aligned(16)));
  __shared__ bf16_t Bs[NBUF * BLK_N * LDB_SW] __attribute__((aligned(16)));
  const int z  = blockIdx.z;
  const int m0 = blockIdx.y * BLK_M;
  const int n0 = blockIdx.x * BLK_N;
  const bf16_t* A = p.A + (size_t)z * p.sA;
  v8f acc[2][4] = {};
  gemm_mainloop(A, p.BT, p.M, CCH, p.K, p.lda, p.ldb, m0, n0, As, Bs, acc);

  const int tid = threadIdx.x, lane = tid & 31, wave = tid >> 5;
  const int wm = wave >> 1, wn = wave & 1;
  const int lr = lane & 15, lh = lane >> 4;
  const int b  = p.c0 + z;
#pragma unroll
  for (int j = 0; j < 4; ++j) {
    const int c = n0 + wn * 64 + j * 16 + lr;   // channel (always < 1024)
    const float wb  = p.w_b[c];
    const float mu  = p.bn_m[c];
    const float inv = p.bn_g[c] * rsqrtf(p.bn_v[c] + 1e-5f);
    const float bb  = p.bn_b[c];
    const float* xrow = p.x + ((size_t)b * CCH + c) * NPOS;
    float sum = 0.f;
#pragma unroll
    for (int i = 0; i < 2; ++i) {
      const int mb = m0 + wm * 32 + i * 16 + lh * 8;
      if (mb + 8 <= p.M) {
        // vectorized residual: rows are 8 consecutive positions
        f32x4 x0 = *(const f32x4*)(xrow + mb);
        f32x4 x1 = *(const f32x4*)(xrow + mb + 4);
#pragma unroll
        for (int r = 0; r < 8; ++r) {
          const float xv = (r < 4) ? x0[r] : x1[r - 4];
          float v = acc[i][j][r] + wb;
          v = (v - mu) * inv + bb + xv;
          sum += v;
        }
      } else if (mb < p.M) {
#pragma unroll
        for (int r = 0; r < 8; ++r) {
          const int m = mb + r;
          if (m < p.M) {
            float v = acc[i][j][r] + wb;
            v = (v - mu) * inv + bb + xrow[m];
            sum += v;
          }
        }
      }
    }
    sum += __shfl_xor(sum, 16, 32);   // fold the two 8-row lane halves
    if (lh == 0)
      atomicAdd(p.pool + (size_t)b * CCH + c, sum * (1.0f / (float)NPOS));
  }
}

// ---------------------------------------------------------------------------
// Small helper kernels
// ---------------------------------------------------------------------------
__global__ void f32_to_bf16_kernel(const float* __restrict__ s, bf16_t* __restrict__ d, int n) {
  int i = blockIdx.x * blockDim.x + threadIdx.x;
  if (i < n) d[i] = (bf16_t)s[i];
}

__global__ void zero_f32_kernel(float* p, int n) {
  int i = blockIdx.x * blockDim.x + threadIdx.x;
  if (i < n) p[i] = 0.f;
}

// x (C x N) f32 per batch -> xT (N x C) bf16 per batch, 32x32 LDS tiles.
// Makes every projection GEMM a pure async-BT consumer.
__global__ __launch_bounds__(256) void transpose_x_kernel(const float* __restrict__ x,
                                                          bf16_t* __restrict__ xT) {
  __shared__ bf16_t tile[32][33];
  const int z  = blockIdx.z;
  const int n0 = blockIdx.x * 32;
  const int c0 = blockIdx.y * 32;
  const int tx = threadIdx.x & 31;
  const int ty = threadIdx.x >> 5;     // 0..7
  const float* xs = x  + (size_t)z * CCH * NPOS;
  bf16_t*      xd = xT + (size_t)z * NPOS * CCH;
#pragma unroll
  for (int r = 0; r < 4; ++r) {
    const int c = c0 + ty * 4 + r;
    const int n = n0 + tx;
    const float v = (n < NPOS) ? xs[(size_t)c * NPOS + n] : 0.f;
    tile[ty * 4 + r][tx] = (bf16_t)v;
  }
  __syncthreads();
#pragma unroll
  for (int r = 0; r < 4; ++r) {
    const int n = n0 + ty * 4 + r;
    const int c = c0 + tx;
    if (n < NPOS) xd[(size_t)n * CCH + c] = tile[tx][ty * 4 + r];
  }
}

// 2x2 spatial max-pool of g/phi projections.
// gfull/phifull: (ICH x NPOS) bf16 per batch.
// gP  : (ICH x NP2P) bf16 pooled, natural layout (BT for the context GEMM)
// phiT: (NP2 x ICH)  bf16 pooled, transposed    (BT for the score GEMM)
__global__ void pool_kernel(const bf16_t* __restrict__ gfull,
                            const bf16_t* __restrict__ phifull,
                            bf16_t* __restrict__ gP, bf16_t* __restrict__ phiT) {
  const int z = blockIdx.y;
  int idx = blockIdx.x * blockDim.x + threadIdx.x;
  if (idx >= ICH * NP2) return;
  const int ic = idx / NP2, n2 = idx % NP2;
  const int t = n2 / 49, rr = n2 % 49, h2 = rr / 7, w2 = rr % 7;
  const int base = t * 196 + (2 * h2) * 14 + 2 * w2;
  const bf16_t* g = gfull   + (size_t)z * ICH * NPOS + (size_t)ic * NPOS;
  const bf16_t* f = phifull + (size_t)z * ICH * NPOS + (size_t)ic * NPOS;
  float mg = fmaxf(fmaxf((float)g[base], (float)g[base + 1]),
                   fmaxf((float)g[base + 14], (float)g[base + 15]));
  float mf = fmaxf(fmaxf((float)f[base], (float)f[base + 1]),
                   fmaxf((float)f[base + 14], (float)f[base + 15]));
  gP  [(size_t)z * ICH * NP2P + (size_t)ic * NP2P + n2] = (bf16_t)mg;
  phiT[(size_t)z * NP2 * ICH + (size_t)n2 * ICH + ic] = (bf16_t)mf;
}

// Row softmax over NP2=196 valid cols (row stride NP2P), writes zero-padded bf16.
// One wave (32 threads) per row.
__global__ void softmax_kernel(const float* __restrict__ S, bf16_t* __restrict__ P) {
  const int row = blockIdx.x, z = blockIdx.y, t = threadIdx.x;
  const float* s = S + ((size_t)z * NPOS + row) * NP2P;
  bf16_t*      p = P + ((size_t)z * NPOS + row) * NP2P;
  float vals[7];
  float mx = -3.0e38f;
#pragma unroll
  for (int j = 0; j < 7; ++j) {
    int i = t + 32 * j;
    float v = (i < NP2) ? s[i] : -3.0e38f;
    vals[j] = v;
    mx = fmaxf(mx, v);
  }
#pragma unroll
  for (int off = 16; off >= 1; off >>= 1) mx = fmaxf(mx, __shfl_xor(mx, off, 32));
  float sum = 0.f;
#pragma unroll
  for (int j = 0; j < 7; ++j) {
    int i = t + 32 * j;
    if (i < NP2) { vals[j] = __expf(vals[j] - mx); sum += vals[j]; }
  }
#pragma unroll
  for (int off = 16; off >= 1; off >>= 1) sum += __shfl_xor(sum, off, 32);
  const float rs = 1.0f / sum;
#pragma unroll
  for (int j = 0; j < 7; ++j) {
    int i = t + 32 * j;
    p[i] = (i < NP2) ? (bf16_t)(vals[j] * rs) : (bf16_t)0.f;
  }
}

// Final FC3: out[b, j] = fc3_b[j] + sum_k concat[b,k] * fc3_w[j,k], concat 2048-wide.
__global__ void fc3_kernel(const float* __restrict__ sb, const float* __restrict__ face1,
                           const float* __restrict__ ib, const float* __restrict__ face2,
                           const float* __restrict__ w, const float* __restrict__ bias,
                           float* __restrict__ out) {
  __shared__ float s0[256], s1[256];
  const int b = blockIdx.x, t = threadIdx.x;
  float a0 = 0.f, a1 = 0.f;
  for (int k = t; k < 2048; k += 256) {
    float v;
    if      (k < 512)  v = sb[(size_t)b * 512 + k];
    else if (k < 1024) v = face1[(size_t)b * 512 + (k - 512)];
    else if (k < 1536) v = ib[(size_t)b * 512 + (k - 1024)];
    else               v = face2[(size_t)b * 512 + (k - 1536)];
    a0 += v * w[k];
    a1 += v * w[2048 + k];
  }
  s0[t] = a0; s1[t] = a1;
  __syncthreads();
  for (int off = 128; off >= 1; off >>= 1) {
    if (t < off) { s0[t] += s0[t + off]; s1[t] += s1[t + off]; }
    __syncthreads();
  }
  if (t == 0) {
    out[b * 2 + 0] = s0[0] + bias[0];
    out[b * 2 + 1] = s1[0] + bias[1];
  }
}

// ---------------------------------------------------------------------------
// Host launcher
// ---------------------------------------------------------------------------
static inline size_t alignup(size_t x) { return (x + 255) & ~(size_t)255; }
static inline dim3 gemm_grid(int M, int N, int Z) {
  return dim3((N + BLK_N - 1) / BLK_N, (M + BLK_M - 1) / BLK_M, Z);
}

extern "C" void kernel_launch(void* const* d_in, const int* in_sizes, int n_in,
                              void* d_out, int out_size, void* d_ws, size_t ws_size,
                              hipStream_t stream) {
  (void)in_sizes; (void)n_in; (void)out_size; (void)ws_size;
  // Input layout (setup_inputs dict order, params flattened in insertion order):
  // 0: x_self_body  1: x_self_face  2: x_interact_body  3: x_interact_face
  // 4+12*blk + {0:g_w 1:g_b 2:theta_w 3:theta_b 4:phi_w 5:phi_b 6:w_w 7:w_b
  //             8:bn_g 9:bn_b 10:bn_m 11:bn_v}  (blk=0 nl1, blk=1 nl2)
  // 28: fc1_w 29: fc1_b 30: fc2_w 31: fc2_b 32: fc3_w 33: fc3_b
  char* w = (char*)d_ws;
  size_t off = 0;
  auto alloc = [&](size_t bytes) -> void* {
    void* p = w + off; off = alignup(off + bytes); return p;
  };
  const size_t WSZ = (size_t)ICH * CCH;  // 512*1024

  bf16_t* wG[2], *wTh[2], *wPhi[2], *wWB[2];
  for (int blk = 0; blk < 2; ++blk) {
    wG[blk]   = (bf16_t*)alloc(WSZ * 2);
    wTh[blk]  = (bf16_t*)alloc(WSZ * 2);
    wPhi[blk] = (bf16_t*)alloc(WSZ * 2);
    wWB[blk]  = (bf16_t*)alloc(WSZ * 2);   // w_w bf16 (C x IC), already BT layout
  }
  bf16_t* fcB[2]   = { (bf16_t*)alloc(WSZ * 2), (bf16_t*)alloc(WSZ * 2) }; // fc_w bf16 (512x1024)
  float*  poolf[2] = { (float*)alloc((size_t)BB * CCH * 4), (float*)alloc((size_t)BB * CCH * 4) };
  bf16_t* poolb[2] = { (bf16_t*)alloc((size_t)BB * CCH * 2), (bf16_t*)alloc((size_t)BB * CCH * 2) };
  float*  sbib[2]  = { (float*)alloc((size_t)BB * ICH * 4), (float*)alloc((size_t)BB * ICH * 4) };

  // chunk scratch
  bf16_t* xbf     = (bf16_t*)alloc((size_t)BCHUNK * NPOS * CCH * 2);  // xT (N x C) bf16
  bf16_t* thetaT  = (bf16_t*)alloc((size_t)BCHUNK * NPOS * ICH * 2);
  bf16_t* gfull   = (bf16_t*)alloc((size_t)BCHUNK * ICH * NPOS * 2);
  bf16_t* phifull = (bf16_t*)alloc((size_t)BCHUNK * ICH * NPOS * 2);
  bf16_t* gP      = (bf16_t*)alloc((size_t)BCHUNK * ICH * NP2P * 2);
  bf16_t* phiT    = (bf16_t*)alloc((size_t)BCHUNK * NP2 * ICH * 2);
  float*  Smat    = (float*)alloc((size_t)BCHUNK * NPOS * NP2P * 4);
  bf16_t* Pmat    = (bf16_t*)alloc((size_t)BCHUNK * NPOS * NP2P * 2);
  bf16_t* Ymat    = (bf16_t*)alloc((size_t)BCHUNK * NPOS * ICH * 2);

  const int CVT = (int)((WSZ + 255) / 256);

  // --- weight conversion to bf16 (BT convention: no weight transposes) ---
  for (int blk = 0; blk < 2; ++blk) {
    const int base = 4 + 12 * blk;
    f32_to_bf16_kernel<<<CVT, 256, 0, stream>>>((const float*)d_in[base + 0], wG[blk],  (int)WSZ);
    f32_to_bf16_kernel<<<CVT, 256, 0, stream>>>((const float*)d_in[base + 2], wTh[blk], (int)WSZ);
    f32_to_bf16_kernel<<<CVT, 256, 0, stream>>>((const float*)d_in[base + 4], wPhi[blk],(int)WSZ);
    f32_to_bf16_kernel<<<CVT, 256, 0, stream>>>((const float*)d_in[base + 6], wWB[blk], (int)WSZ);
  }
  f32_to_bf16_kernel<<<CVT, 256, 0, stream>>>((const float*)d_in[28], fcB[0], (int)WSZ);
  f32_to_bf16_kernel<<<CVT, 256, 0, stream>>>((const float*)d_in[30], fcB[1], (int)WSZ);
  zero_f32_kernel<<<(BB * CCH + 255) / 256, 256, 0, stream>>>(poolf[0], BB * CCH);
  zero_f32_kernel<<<(BB * CCH + 255) / 256, 256, 0, stream>>>(poolf[1], BB * CCH);

  // --- non-local blocks, batch chunks ---
  for (int c0 = 0; c0 < BB; c0 += BCHUNK) {
    for (int blk = 0; blk < 2; ++blk) {
      const float* xb = (const float*)d_in[blk == 0 ? 0 : 2];
      const int base = 4 + 12 * blk;
      const float* xchunk = xb + (size_t)c0 * CCH * NPOS;

      // x chunk -> xT bf16 (one pass; all GEMMs below are async-BT consumers)
      transpose_x_kernel<<<dim3((NPOS + 31) / 32, CCH / 32, BCHUNK), 256, 0, stream>>>(xchunk, xbf);

      GemmP p{};
      // theta: (IC x C)*(C x N) -> thetaT (N x IC) bf16, bias per row (channel)
      p.A = wTh[blk]; p.BT = xbf; p.bias = (const float*)d_in[base + 3];
      p.D = thetaT; p.sA = 0; p.sB = (long)NPOS * CCH; p.sD = (long)NPOS * ICH;
      p.lda = CCH; p.ldb = CCH; p.ldd = ICH; p.M = ICH; p.N = NPOS; p.K = CCH;
      gemm_bf16_kernel<2, 1><<<gemm_grid(ICH, NPOS, BCHUNK), 256, 0, stream>>>(p);
      // g: -> gfull (IC x N) bf16
      p.A = wG[blk]; p.bias = (const float*)d_in[base + 1];
      p.D = gfull; p.sD = (long)ICH * NPOS; p.ldd = NPOS;
      gemm_bf16_kernel<1, 1><<<gemm_grid(ICH, NPOS, BCHUNK), 256, 0, stream>>>(p);
      // phi: -> phifull (IC x N) bf16
      p.A = wPhi[blk]; p.bias = (const float*)d_in[base + 5];
      p.D = phifull;
      gemm_bf16_kernel<1, 1><<<gemm_grid(ICH, NPOS, BCHUNK), 256, 0, stream>>>(p);

      // 2x2 max-pool -> gP (IC x NP2P), phiT (NP2 x IC)
      pool_kernel<<<dim3((ICH * NP2 + 255) / 256, BCHUNK), 256, 0, stream>>>(gfull, phifull, gP, phiT);

      // scores: thetaT (N x IC) x phi (IC x N2), BT = phiT -> S f32 (N x NP2P)
      GemmP ps{};
      ps.A = thetaT; ps.BT = phiT; ps.bias = nullptr; ps.D = Smat;
      ps.sA = (long)NPOS * ICH; ps.sB = (long)NP2 * ICH; ps.sD = (long)NPOS * NP2P;
      ps.lda = ICH; ps.ldb = ICH; ps.ldd = NP2P; ps.M = NPOS; ps.N = NP2; ps.K = ICH;
      gemm_bf16_kernel<0, 0><<<gemm_grid(NPOS, NP2, BCHUNK), 256, 0, stream>>>(ps);

      softmax_kernel<<<dim3(NPOS, BCHUNK), 32, 0, stream>>>(Smat, Pmat);

      // context: P (N x NP2P) x g (NP2P x IC), BT = gP -> Y (N x IC) bf16
      GemmP py{};
      py.A = Pmat; py.BT = gP; py.bias = nullptr; py.D = Ymat;
      py.sA = (long)NPOS * NP2P; py.sB = (long)ICH * NP2P; py.sD = (long)NPOS * ICH;
      py.lda = NP2P; py.ldb = NP2P; py.ldd = ICH; py.M = NPOS; py.N = ICH; py.K = NP2P;
      gemm_bf16_kernel<1, 0><<<gemm_grid(NPOS, ICH, BCHUNK), 256, 0, stream>>>(py);

      // z = Y x w_w^T (BT = w_w), fused bias+BN+residual+mean-pool
      GemmZP pz{};
      pz.A = Ymat; pz.BT = wWB[blk];
      pz.w_b  = (const float*)d_in[base + 7];
      pz.bn_g = (const float*)d_in[base + 8];
      pz.bn_b = (const float*)d_in[base + 9];
      pz.bn_m = (const float*)d_in[base + 10];
      pz.bn_v = (const float*)d_in[base + 11];
      pz.x = xb; pz.pool = poolf[blk];
      pz.sA = (long)NPOS * ICH; pz.lda = ICH; pz.ldb = ICH;
      pz.M = NPOS; pz.K = ICH; pz.c0 = c0;
      gemm_z_kernel<<<gemm_grid(NPOS, CCH, BCHUNK), 256, 0, stream>>>(pz);
    }
  }

  // --- FC heads ---
  f32_to_bf16_kernel<<<(BB * CCH + 255) / 256, 256, 0, stream>>>(poolf[0], poolb[0], BB * CCH);
  f32_to_bf16_kernel<<<(BB * CCH + 255) / 256, 256, 0, stream>>>(poolf[1], poolb[1], BB * CCH);
  for (int i = 0; i < 2; ++i) {
    GemmP pf{};
    pf.A = poolb[i]; pf.BT = fcB[i];     // BT = fc_w (512 x 1024)
    pf.bias = (const float*)d_in[i == 0 ? 29 : 31];
    pf.D = sbib[i];
    pf.sA = 0; pf.sB = 0; pf.sD = 0;
    pf.lda = CCH; pf.ldb = CCH; pf.ldd = ICH; pf.M = BB; pf.N = ICH; pf.K = CCH;
    gemm_bf16_kernel<0, 2><<<gemm_grid(BB, ICH, 1), 256, 0, stream>>>(pf);
  }
  fc3_kernel<<<BB, 256, 0, stream>>>(sbib[0], (const float*)d_in[1],
                                     sbib[1], (const float*)d_in[3],
                                     (const float*)d_in[32], (const float*)d_in[33],
                                     (float*)d_out);
}
